// MultiQueryAttention_31104153157879
// MI455X (gfx1250) — matrix operand order repaired and checked by reading, not compiled, not run on hardware
//
#include <hip/hip_runtime.h>
#include <hip/hip_bf16.h>

// ---------------------------------------------------------------------------
// MQA: B=4, S=2048, N=2048, D=1024, H=16, HD=64.  Flash attention + WMMA bf16.
// ---------------------------------------------------------------------------

typedef __attribute__((ext_vector_type(16))) __bf16 v16bf;
typedef __attribute__((ext_vector_type(8)))  __bf16 v8bf;
typedef __attribute__((ext_vector_type(8)))  float  v8f;
typedef int gv4i __attribute__((vector_size(16)));   // matches builtin's V4i

#define BB   4
#define SS   2048
#define NNK  2048
#define DDIM 1024
#define HH   16
#define HDIM 64
#define NJ   (NNK / 32)        // 64 key chunks of 32
#define EPSLN 1e-5f
#define LOG2E 1.4426950408889634f

#if defined(__HIP_DEVICE_COMPILE__) && \
    __has_builtin(__builtin_amdgcn_global_load_async_to_lds_b128) && \
    __has_builtin(__builtin_amdgcn_s_wait_asynccnt)
#define USE_ASYNC 1
#else
#define USE_ASYNC 0
#endif

__device__ __forceinline__ void stage16(const __bf16* g, __bf16* l) {
#if USE_ASYNC
  __builtin_amdgcn_global_load_async_to_lds_b128(
      (__attribute__((address_space(1))) gv4i*)g,
      (__attribute__((address_space(3))) gv4i*)l, 0, 0);
#else
  *(uint4*)l = *(const uint4*)g;
#endif
}

// max-butterfly within 16-lane rows via DPP ROW_XMASK (0x160|mask): pure VALU,
// no DS traffic (vs ds_bpermute that __shfl_xor lowers to).
template <int CTRL>
__device__ __forceinline__ float xmask_max(float x) {
  const int yi = __builtin_amdgcn_update_dpp(0, __float_as_int(x), CTRL, 0xf, 0xf, true);
  return fmaxf(x, __int_as_float(yi));
}
__device__ __forceinline__ float rowmax16(float x) {
  x = xmask_max<0x161>(x);
  x = xmask_max<0x162>(x);
  x = xmask_max<0x164>(x);
  x = xmask_max<0x168>(x);
  return x;
}

// ---------------------------------------------------------------------------
// Kernel 1: layernorm K -> bf16 (row-major), V -> bf16 transposed per 32-key
// chunk: Vt[b][chunk j][feature f (64)][key kk (32)].
// One wave per row; 8 waves/block; grid = B*N/8 = 1024.
// ---------------------------------------------------------------------------
__global__ __launch_bounds__(256) void prep_kv(
    const float* __restrict__ xk, const float* __restrict__ xv,
    const float* __restrict__ knw, const float* __restrict__ knb,
    __bf16* __restrict__ Kbf, __bf16* __restrict__ Vt) {
  const int wave = threadIdx.x >> 5, lane = threadIdx.x & 31;
  const int r = blockIdx.x * 8 + wave;          // 0 .. B*N-1
  const int b = r >> 11, n = r & 2047;
  const float* krow = xk + ((size_t)r << 6);
  const int f0 = lane * 2;
  float x0 = krow[f0], x1 = krow[f0 + 1];
  float s1 = x0 + x1, s2 = x0 * x0 + x1 * x1;
#pragma unroll
  for (int off = 1; off < 32; off <<= 1) {
    s1 += __shfl_xor(s1, off);
    s2 += __shfl_xor(s2, off);
  }
  const float mean = s1 * (1.0f / 64.0f);
  const float rstd = rsqrtf(s2 * (1.0f / 64.0f) - mean * mean + EPSLN);
  Kbf[((size_t)r << 6) + f0]     = (__bf16)((x0 - mean) * rstd * knw[f0] + knb[f0]);
  Kbf[((size_t)r << 6) + f0 + 1] = (__bf16)((x1 - mean) * rstd * knw[f0 + 1] + knb[f0 + 1]);
  // V transpose into chunk-major layout
  const float* vrow = xv + ((size_t)r << 6);
  float v0 = vrow[f0], v1 = vrow[f0 + 1];
  const int j = n >> 5, kk = n & 31;
  __bf16* vt = Vt + ((size_t)(b * NJ + j)) * (64 * 32);
  vt[(size_t)f0 * 32 + kk]       = (__bf16)v0;
  vt[(size_t)(f0 + 1) * 32 + kk] = (__bf16)v1;
}

// ---------------------------------------------------------------------------
// Kernel 2: flash attention.  8 waves/block, 16 q-rows per wave (128/block),
// grid = B*H*S/128 = 1024.  Q layernorm+scale fused; K/V chunks staged to LDS
// (double buffered, async); online softmax in exp2 domain; row sums of P via
// an extra WMMA against a ones matrix (C-layout hands each lane its rows).
// ---------------------------------------------------------------------------
__global__ __launch_bounds__(256) void mqa_flash(
    const float* __restrict__ xq, const __bf16* __restrict__ Kbf,
    const __bf16* __restrict__ Vt, const float* __restrict__ qnw,
    const float* __restrict__ qnb, float* __restrict__ Obuf) {
  __shared__ __align__(32) __bf16 sK[2][32 * 64];   // key-major (key, feat)
  __shared__ __align__(32) __bf16 sV[2][64 * 32];   // feat-major (feat, key)
  __shared__ __align__(32) __bf16 sP[8][16 * 32];   // per-wave P bounce

  const int tid = threadIdx.x;
  const int wave = tid >> 5, lane = tid & 31;
  const int lane16 = lane & 15, hi = lane >> 4;
  const int grow_blk = blockIdx.x * 128;
  const int b = grow_blk >> 15;                       // / (H*S)
  const int myrow = grow_blk + wave * 16 + lane16;    // q row for this lane-pair
  const int base = hi * 16;                           // A-fragment K offset

  // ---- Q layernorm + scale, build A fragments in WMMA lane layout ----
  const float* qrow = xq + ((size_t)myrow << 6);
  float qv[32];
  float s1 = 0.f, s2 = 0.f;
#pragma unroll
  for (int g = 0; g < 4; ++g) {
    const float4* p = (const float4*)(qrow + base + (g & 1) * 8 + (g >> 1) * 32);
    float4 u0 = p[0], u1 = p[1];
    qv[g * 8 + 0] = u0.x; qv[g * 8 + 1] = u0.y; qv[g * 8 + 2] = u0.z; qv[g * 8 + 3] = u0.w;
    qv[g * 8 + 4] = u1.x; qv[g * 8 + 5] = u1.y; qv[g * 8 + 6] = u1.z; qv[g * 8 + 7] = u1.w;
    s1 += u0.x + u0.y + u0.z + u0.w + u1.x + u1.y + u1.z + u1.w;
    s2 += u0.x * u0.x + u0.y * u0.y + u0.z * u0.z + u0.w * u0.w +
          u1.x * u1.x + u1.y * u1.y + u1.z * u1.z + u1.w * u1.w;
  }
  s1 += __shfl_xor(s1, 16);
  s2 += __shfl_xor(s2, 16);
  const float mean = s1 * (1.0f / 64.0f);
  const float rstd = rsqrtf(s2 * (1.0f / 64.0f) - mean * mean + EPSLN);
  const float QSC = 0.125f * LOG2E;   // HD^-0.5 * log2(e): softmax in exp2 domain
  v16bf a0, a1;
#pragma unroll
  for (int g = 0; g < 4; ++g) {
#pragma unroll
    for (int jj = 0; jj < 8; ++jj) {
      const int f = base + (g & 1) * 8 + (g >> 1) * 32 + jj;
      const float y = ((qv[g * 8 + jj] - mean) * rstd * qnw[f] + qnb[f]) * QSC;
      if (g < 2) a0[g * 8 + jj] = (__bf16)y;
      else       a1[(g - 2) * 8 + jj] = (__bf16)y;
    }
  }

  // all-ones B fragment for WMMA row sums of P
  v16bf onesb;
#pragma unroll
  for (int e = 0; e < 16; ++e) onesb[e] = (__bf16)1.0f;

  const __bf16* kgb = Kbf + ((size_t)b * NNK) * 64;
  const __bf16* vgb = Vt + ((size_t)b * NJ) * (64 * 32);
  auto issue = [&](int j, int buf) {
    stage16(kgb + (size_t)j * (32 * 64) + tid * 8, &sK[buf][tid * 8]);
    stage16(vgb + (size_t)j * (64 * 32) + tid * 8, &sV[buf][tid * 8]);
  };

  float m_i[8], l_i[8];
  v8f o[4];
  const v8f vzero = {0.f, 0.f, 0.f, 0.f, 0.f, 0.f, 0.f, 0.f};
#pragma unroll
  for (int i = 0; i < 8; ++i) { m_i[i] = -3.0e38f; l_i[i] = 0.f; }
#pragma unroll
  for (int t = 0; t < 4; ++t) o[t] = vzero;

  issue(0, 0);
  for (int j = 0; j < NJ; ++j) {
    const int cur = j & 1;
#if USE_ASYNC
    __builtin_amdgcn_s_wait_asynccnt(0);
#endif
    __syncthreads();
    if (j + 1 < NJ) issue(j + 1, cur ^ 1);

    // ---- S = Q * K^T : 2 n-tiles x 2 k-steps = 4 WMMAs ----
    const __bf16* kc = &sK[cur][0];
    v8f sc[2];
#pragma unroll
    for (int t = 0; t < 2; ++t) {
      const __bf16* kr = kc + (t * 16 + lane16) * 64;
      v16bf b0 = *(const v16bf*)(kr + hi * 16);
      v16bf b1 = *(const v16bf*)(kr + 32 + hi * 16);
      v8f c = __builtin_amdgcn_wmma_f32_16x16x32_bf16(false, a0, false, b0,
                                                      (short)0, vzero, false, false);
      c = __builtin_amdgcn_wmma_f32_16x16x32_bf16(false, a1, false, b1,
                                                  (short)0, c, false, false);
      sc[t] = c;
    }

    // ---- online softmax: DPP max butterflies, exp2 domain ----
    float p0[8], p1[8], alpha[8];
#pragma unroll
    for (int i = 0; i < 8; ++i) {
      const float cm = rowmax16(fmaxf(sc[0][i], sc[1][i]));
      const float mn = fmaxf(m_i[i], cm);
      alpha[i] = exp2f(m_i[i] - mn);
      m_i[i] = mn;
      p0[i] = exp2f(sc[0][i] - mn);
      p1[i] = exp2f(sc[1][i] - mn);
    }
#pragma unroll
    for (int t = 0; t < 4; ++t)
#pragma unroll
      for (int i = 0; i < 8; ++i) o[t][i] *= alpha[i];

    // ---- bounce P through LDS: C-layout -> A-layout, as bf16 ----
    __bf16* ps = &sP[wave][0];
#pragma unroll
    for (int i = 0; i < 8; ++i) {
      const int m = hi * 8 + i;
      ps[m * 32 + lane16]      = (__bf16)p0[i];
      ps[m * 32 + 16 + lane16] = (__bf16)p1[i];
    }
    __syncthreads();
    v8bf pl0 = *(const v8bf*)(ps + lane16 * 32 + base);
    v8bf pl1 = *(const v8bf*)(ps + lane16 * 32 + base + 8);
    v16bf pa;
#pragma unroll
    for (int jj = 0; jj < 8; ++jj) { pa[jj] = pl0[jj]; pa[jj + 8] = pl1[jj]; }

    // ---- row sums of P via one WMMA against ones(32x16) ----
    v8f srow = __builtin_amdgcn_wmma_f32_16x16x32_bf16(false, pa, false, onesb,
                                                       (short)0, vzero, false, false);
#pragma unroll
    for (int i = 0; i < 8; ++i) l_i[i] = l_i[i] * alpha[i] + srow[i];

    // ---- O += P * V : 4 n-tiles (HD=64), K=32 keys ----
    const __bf16* vc = &sV[cur][0];
#pragma unroll
    for (int t = 0; t < 4; ++t) {
      v16bf vb = *(const v16bf*)(vc + (t * 16 + lane16) * 32 + hi * 16);
      o[t] = __builtin_amdgcn_wmma_f32_16x16x32_bf16(false, pa, false, vb,
                                                     (short)0, o[t], false, false);
    }
    __syncthreads();
  }

  // ---- finalize: divide by softmax denominator, store (B*H*S, 64) fp32 ----
#pragma unroll
  for (int i = 0; i < 8; ++i) {
    const float inv = 1.0f / l_i[i];
    const int m = hi * 8 + i;
    float* orow = Obuf + ((size_t)(grow_blk + wave * 16 + m) << 6);
#pragma unroll
    for (int t = 0; t < 4; ++t) orow[t * 16 + lane16] = o[t][i] * inv;
  }
}

// ---------------------------------------------------------------------------
// Kernel 3a: gather-transpose (B,H,S,HD)->(B,S,D), layernorm over D, -> bf16.
// One block per (b, s) row; 256 threads x 4 elements.  grid = B*S = 8192.
// ---------------------------------------------------------------------------
__global__ __launch_bounds__(256) void ln_gather(
    const float* __restrict__ Obuf, const float* __restrict__ nw,
    const float* __restrict__ nb, __bf16* __restrict__ Xln) {
  __shared__ float r1[8], r2[8];
  __shared__ float bmean, brstd;
  const int row = blockIdx.x;
  const int b = row >> 11, s2 = row & 2047;
  const int t = threadIdx.x, lane = t & 31, wave = t >> 5;
  const int d = t * 4, h = d >> 6, hd = d & 63;
  const float* src = Obuf + (((size_t)(b * HH + h) * SS + s2) << 6) + hd;
  const float4 v = *(const float4*)src;
  float s1 = v.x + v.y + v.z + v.w;
  float s2v = v.x * v.x + v.y * v.y + v.z * v.z + v.w * v.w;
#pragma unroll
  for (int off = 1; off < 32; off <<= 1) {
    s1 += __shfl_xor(s1, off);
    s2v += __shfl_xor(s2v, off);
  }
  if (lane == 0) { r1[wave] = s1; r2[wave] = s2v; }
  __syncthreads();
  if (t == 0) {
    float a = 0.f, c = 0.f;
#pragma unroll
    for (int w = 0; w < 8; ++w) { a += r1[w]; c += r2[w]; }
    const float mn = a * (1.0f / 1024.0f);
    bmean = mn;
    brstd = rsqrtf(c * (1.0f / 1024.0f) - mn * mn + EPSLN);
  }
  __syncthreads();
  const float mn = bmean, rs = brstd;
  const float4 wv = *(const float4*)(nw + d);
  const float4 bv = *(const float4*)(nb + d);
  __bf16* dst = Xln + ((size_t)row << 10) + d;
  dst[0] = (__bf16)((v.x - mn) * rs * wv.x + bv.x);
  dst[1] = (__bf16)((v.y - mn) * rs * wv.y + bv.y);
  dst[2] = (__bf16)((v.z - mn) * rs * wv.z + bv.z);
  dst[3] = (__bf16)((v.w - mn) * rs * wv.w + bv.w);
}

// Kernel 3w: convert proj_w (D,D) fp32 -> bf16 (layout already B-fragment ok).
__global__ __launch_bounds__(256) void conv_w(const float* __restrict__ W,
                                              __bf16* __restrict__ Wb) {
  const int i = (blockIdx.x * 256 + threadIdx.x) * 4;
  const float4 v = *(const float4*)(W + i);
  Wb[i] = (__bf16)v.x; Wb[i + 1] = (__bf16)v.y;
  Wb[i + 2] = (__bf16)v.z; Wb[i + 3] = (__bf16)v.w;
}

// ---------------------------------------------------------------------------
// Kernel 3b: out = Xln(8192x1024) * W^T(1024x1024), bf16 WMMA, fp32 out.
// One 16x16 tile per wave, K-loop of 32 steps.  grid = 8192/16*64/8 = 4096.
// ---------------------------------------------------------------------------
__global__ __launch_bounds__(256) void proj_gemm(
    const __bf16* __restrict__ Xln, const __bf16* __restrict__ Wb,
    float* __restrict__ out) {
  const int tid = threadIdx.x, wave = tid >> 5, lane = tid & 31;
  const int lane16 = lane & 15, hi = lane >> 4;
  const int tile = blockIdx.x * 8 + wave;
  const int rt = tile >> 6, ct = tile & 63;
  const __bf16* arow = Xln + ((size_t)(rt * 16 + lane16) << 10);
  const __bf16* brow = Wb + ((size_t)(ct * 16 + lane16) << 10);
  const v8f vzero = {0.f, 0.f, 0.f, 0.f, 0.f, 0.f, 0.f, 0.f};
  v8f acc = vzero;
#pragma unroll 4
  for (int ks = 0; ks < 32; ++ks) {
    v8bf x0 = *(const v8bf*)(arow + ks * 32 + hi * 16);
    v8bf x1 = *(const v8bf*)(arow + ks * 32 + hi * 16 + 8);
    v16bf af;
#pragma unroll
    for (int jj = 0; jj < 8; ++jj) { af[jj] = x0[jj]; af[jj + 8] = x1[jj]; }
    v16bf bf_ = *(const v16bf*)(brow + ks * 32 + hi * 16);
    acc = __builtin_amdgcn_wmma_f32_16x16x32_bf16(false, af, false, bf_,
                                                  (short)0, acc, false, false);
  }
#pragma unroll
  for (int i = 0; i < 8; ++i)
    out[((size_t)(rt * 16 + hi * 8 + i) << 10) + ct * 16 + lane16] = acc[i];
}

// ---------------------------------------------------------------------------
extern "C" void kernel_launch(void* const* d_in, const int* in_sizes, int n_in,
                              void* d_out, int out_size, void* d_ws, size_t ws_size,
                              hipStream_t stream) {
  (void)in_sizes; (void)n_in; (void)out_size; (void)ws_size;
  const float* xq  = (const float*)d_in[0];
  const float* xk  = (const float*)d_in[1];
  const float* xv  = (const float*)d_in[2];
  const float* qnw = (const float*)d_in[3];
  const float* qnb = (const float*)d_in[4];
  const float* knw = (const float*)d_in[5];
  const float* knb = (const float*)d_in[6];
  const float* nw  = (const float*)d_in[7];
  const float* nb  = (const float*)d_in[8];
  const float* pw  = (const float*)d_in[9];

  char* ws = (char*)d_ws;
  __bf16* Kbf  = (__bf16*)(ws);                                   // 1 MB
  __bf16* Vt   = (__bf16*)(ws + (size_t)(1u << 20));              // 1 MB
  float*  Obuf = (float*)(ws + (size_t)(2u << 20));               // 32 MB
  __bf16* Xln  = (__bf16*)(ws + (size_t)(2u << 20) + (size_t)(32u << 20));  // 16 MB
  __bf16* Wb   = (__bf16*)(ws + (size_t)(2u << 20) + (size_t)(48u << 20));  // 2 MB

  hipLaunchKernelGGL(prep_kv,   dim3(1024), dim3(256), 0, stream, xk, xv, knw, knb, Kbf, Vt);
  hipLaunchKernelGGL(conv_w,    dim3(1024), dim3(256), 0, stream, pw, Wb);
  hipLaunchKernelGGL(mqa_flash, dim3(1024), dim3(256), 0, stream, xq, Kbf, Vt, qnw, qnb, Obuf);
  hipLaunchKernelGGL(ln_gather, dim3(8192), dim3(256), 0, stream, Obuf, nw, nb, Xln);
  hipLaunchKernelGGL(proj_gemm, dim3(4096), dim3(256), 0, stream, Xln, Wb, (float*)d_out);
}
